// MSHGNN_65970697667190
// MI455X (gfx1250) — matrix operand order repaired and checked by reading, not compiled, not run on hardware
//
#include <hip/hip_runtime.h>
#include <hip/hip_fp16.h>

// ---------------------------------------------------------------------------
// MSHGNN forward for MI455X (gfx1250, wave32).
//  - 5 GEMMs [NS,64]@[64,512] via v_wmma_f32_16x16x32_f16 (1 wave = 1 16x16 tile)
//    (4 projection GEMMs + 1 fused-residual GEMM: sum_g x@Wres_g = x@(sum Wres_g))
//  - edge softmax via encoded-u32 atomicMax + f32 atomicAdd (L2-resident, 192MB)
//  - scatter: 1 wave per (edge, head), 2 floats per lane
// ---------------------------------------------------------------------------

typedef __attribute__((ext_vector_type(16))) _Float16 v16h;
typedef __attribute__((ext_vector_type(8)))  _Float16 v8h;
typedef __attribute__((ext_vector_type(8)))  float    v8f;

#define BT 256            // 8 waves per block
#define NCOL 512          // H*D
#define HH 8
#define DD 64
#define SLOPE 0.2f

// ---------------------------- helpers --------------------------------------

__device__ __forceinline__ float lrelu(float x) { return x > 0.f ? x : SLOPE * x; }

// order-preserving float->u32 encoding (monotonic: a<b  <=>  enc(a)<enc(b))
__device__ __forceinline__ unsigned fenc(float x) {
    unsigned b = __float_as_uint(x);
    return (b & 0x80000000u) ? ~b : (b | 0x80000000u);
}
__device__ __forceinline__ float fdec(unsigned k) {
    unsigned b = (k & 0x80000000u) ? (k ^ 0x80000000u) : ~k;
    return __uint_as_float(b);
}

// ---------------------------- converts -------------------------------------

__global__ void k_cvt_f16(const float* __restrict__ src, _Float16* __restrict__ dst, int n) {
    int i = blockIdx.x * blockDim.x + threadIdx.x;
    if (i < n) dst[i] = (_Float16)src[i];
}

// W [64,512] row-major (f32)  ->  Wt [512,64] (f16)  so B-operand loads are contiguous
__global__ void k_cvt_transpose(const float* __restrict__ W, _Float16* __restrict__ Wt) {
    int i = blockIdx.x * blockDim.x + threadIdx.x;
    if (i < 64 * NCOL) {
        int k = i / NCOL, n = i % NCOL;
        Wt[n * 64 + k] = (_Float16)W[i];
    }
}

// sum of 4 residual weights (f32) -> transposed f16: one GEMM replaces four
__global__ void k_cvt_transpose_sum4(const float* __restrict__ W0, const float* __restrict__ W1,
                                     const float* __restrict__ W2, const float* __restrict__ W3,
                                     _Float16* __restrict__ Wt) {
    int i = blockIdx.x * blockDim.x + threadIdx.x;
    if (i < 64 * NCOL) {
        int k = i / NCOL, n = i % NCOL;
        Wt[n * 64 + k] = (_Float16)(W0[i] + W1[i] + W2[i] + W3[i]);
    }
}

// ---------------------------- WMMA GEMM ------------------------------------
// Out[M,512] = X[M,64] @ Wt^T   (Wt stored [512,64], f16; f32 accumulate)
__global__ void k_gemm_wmma(const _Float16* __restrict__ X, const _Float16* __restrict__ Wt,
                            float* __restrict__ Out, int M) {
    int gid  = blockIdx.x * blockDim.x + threadIdx.x;
    int wave = gid >> 5;
    int lane = gid & 31;
    int tileN = wave & 31;        // 512/16 = 32 N-tiles
    int tileM = wave >> 5;
    if (tileM * 16 >= M) return;  // wave-uniform: EXEC stays all-ones for WMMA

    int hi  = lane >> 4;          // 0: lanes 0-15, 1: lanes 16-31
    int l15 = lane & 15;
    const _Float16* xrow = X  + (size_t)(tileM * 16 + l15) * 64;  // A row M=l15
    const _Float16* wrow = Wt + (size_t)(tileN * 16 + l15) * 64;  // B col N=l15

    v8f c = {};
#pragma unroll
    for (int ko = 0; ko < 2; ++ko) {
        // A layout: elem j -> K = ko*32 + hi*8 + (j<8 ? j : 16+j-8)
        int ka = ko * 32 + hi * 8;
        v8h a0 = *(const v8h*)(xrow + ka);
        v8h a1 = *(const v8h*)(xrow + ka + 16);
        // B layout: elem j -> K = ko*32 + hi*8 + (j<8 ? j : 16+j-8)
        int kb = ko * 32 + hi * 8;
        v8h b0 = *(const v8h*)(wrow + kb);
        v8h b1 = *(const v8h*)(wrow + kb + 16);
        v16h a, b;
#pragma unroll
        for (int j = 0; j < 8; ++j) { a[j] = a0[j]; a[j + 8] = a1[j]; }
#pragma unroll
        for (int j = 0; j < 8; ++j) { b[j] = b0[j]; b[j + 8] = b1[j]; }
        c = __builtin_amdgcn_wmma_f32_16x16x32_f16(false, a, false, b,
                                                   (short)0, c, false, false);
    }
    // D layout: VGPR j at lane -> row (hi*8+j), col l15; straight stores, no RMW
    float* obase = Out + (size_t)(tileM * 16 + hi * 8) * NCOL + tileN * 16 + l15;
#pragma unroll
    for (int j = 0; j < 8; ++j)
        obase[(size_t)j * NCOL] = c[j];
}

// ---------------------------- attention ------------------------------------

// el[n,h] = sum_d h[n,h,d]*al[h,d] ; er likewise
__global__ void k_attn_coef(const float* __restrict__ hbuf, const float* __restrict__ al,
                            const float* __restrict__ ar, float* __restrict__ el,
                            float* __restrict__ er, int NSn) {
    int i = blockIdx.x * blockDim.x + threadIdx.x;
    if (i >= NSn * HH) return;
    int n = i >> 3, hd = i & 7;
    const float* hrow = hbuf + (size_t)n * NCOL + hd * DD;
    const float* pa = al + hd * DD;
    const float* pr = ar + hd * DD;
    float sl = 0.f, sr = 0.f;
#pragma unroll 8
    for (int d = 0; d < DD; ++d) { float hv = hrow[d]; sl += hv * pa[d]; sr += hv * pr[d]; }
    el[i] = sl; er[i] = sr;
}

__global__ void k_init_ms(unsigned* __restrict__ menc, float* __restrict__ s, int n) {
    int i = blockIdx.x * blockDim.x + threadIdx.x;
    if (i < n) { menc[i] = 0u; s[i] = 0.f; }   // key 0 < enc(any finite) -> marks empty
}

__global__ void k_edge_max(const int* __restrict__ U, const int* __restrict__ V,
                           const float* __restrict__ el, const float* __restrict__ er,
                           unsigned* __restrict__ menc, int E) {
    int i = blockIdx.x * blockDim.x + threadIdx.x;
    if (i >= E * HH) return;
    int e = i >> 3, hd = i & 7;
    int u = U[e], v = V[e];
    float a = lrelu(el[u * HH + hd] + er[v * HH + hd]);
    atomicMax(&menc[v * HH + hd], fenc(a));
}

__global__ void k_decode_m(const unsigned* __restrict__ menc, float* __restrict__ m, int n) {
    int i = blockIdx.x * blockDim.x + threadIdx.x;
    if (i < n) { unsigned k = menc[i]; m[i] = (k == 0u) ? 0.f : fdec(k); }  // empty seg -> 0
}

__global__ void k_edge_sum(const int* __restrict__ U, const int* __restrict__ V,
                           const float* __restrict__ el, const float* __restrict__ er,
                           const float* __restrict__ m, float* __restrict__ s, int E) {
    int i = blockIdx.x * blockDim.x + threadIdx.x;
    if (i >= E * HH) return;
    int e = i >> 3, hd = i & 7;
    int u = U[e], v = V[e];
    int vi = v * HH + hd;
    float a = lrelu(el[u * HH + hd] + er[vi]);
    atomicAdd(&s[vi], __expf(a - m[vi]));
}

// 1 wave per (edge, head); lane covers d and d+32. acc/hbuf are L2-resident (2x100MB < 192MB)
__global__ void k_edge_scatter(const int* __restrict__ U, const int* __restrict__ V,
                               const float* __restrict__ el, const float* __restrict__ er,
                               const float* __restrict__ m, const float* __restrict__ s,
                               const float* __restrict__ hbuf, float* __restrict__ acc, int E) {
    int gid  = blockIdx.x * blockDim.x + threadIdx.x;
    int wave = gid >> 5;
    int lane = gid & 31;
    if (wave >= E * HH) return;
    int e = wave >> 3, hd = wave & 7;
    int u = U[e], v = V[e];
    int vi = v * HH + hd;
    float a = lrelu(el[u * HH + hd] + er[vi]);
    float alpha = __expf(a - m[vi]) / s[vi];
    const float* hrow = hbuf + (size_t)u * NCOL + hd * DD;
    float* arow = acc + (size_t)v * NCOL + hd * DD;
    atomicAdd(&arow[lane],      alpha * hrow[lane]);
    atomicAdd(&arow[lane + 32], alpha * hrow[lane + 32]);
}

// ---------------------------- epilogues ------------------------------------

// segment mean over n_per rows then broadcast back (f1 / c1 outputs)
__global__ void k_bcast_mean(const float* __restrict__ x, float* __restrict__ out,
                             int Bn, int nper) {
    int i = blockIdx.x * blockDim.x + threadIdx.x;
    if (i >= Bn * DD) return;
    int b = i >> 6, d = i & 63;
    const float* base = x + (size_t)b * nper * DD + d;
    float sum = 0.f;
    for (int r = 0; r < nper; ++r) sum += base[(size_t)r * DD];
    float mv = sum / (float)nper;
    float* ob = out + (size_t)b * nper * DD + d;
    for (int r = 0; r < nper; ++r) ob[(size_t)r * DD] = mv;
}

// per-graph mean only (for s1, consumed by finalize)
__global__ void k_seg_mean(const float* __restrict__ x, float* __restrict__ mean,
                           int Bn, int nper) {
    int i = blockIdx.x * blockDim.x + threadIdx.x;
    if (i >= Bn * DD) return;
    int b = i >> 6, d = i & 63;
    const float* base = x + (size_t)b * nper * DD + d;
    float sum = 0.f;
    for (int r = 0; r < nper; ++r) sum += base[(size_t)r * DD];
    mean[i] = sum / (float)nper;
}

__global__ void k_bias_sum(const float* __restrict__ b0, const float* __restrict__ b1,
                           const float* __restrict__ b2, const float* __restrict__ b3,
                           float* __restrict__ bsum) {
    int i = blockIdx.x * blockDim.x + threadIdx.x;
    if (i < NCOL) bsum[i] = b0[i] + b1[i] + b2[i] + b3[i];
}

// h_s1[n,d] = max_h (acc[n,h,d] + bsum[h,d]) + graph_mean[n/nper, d]
__global__ void k_finalize_s1(const float* __restrict__ acc, const float* __restrict__ bsum,
                              const float* __restrict__ mean_s1, float* __restrict__ out,
                              int NSn, int nper) {
    int i = blockIdx.x * blockDim.x + threadIdx.x;
    if (i >= NSn * DD) return;
    int n = i >> 6, d = i & 63;
    const float* arow = acc + (size_t)n * NCOL + d;
    float v = -3.402823466e38f;
#pragma unroll
    for (int hd = 0; hd < HH; ++hd)
        v = fmaxf(v, arow[hd * DD] + bsum[hd * DD + d]);
    out[i] = v + mean_s1[(n / nper) * DD + d];
}

// ---------------------------- launcher -------------------------------------

extern "C" void kernel_launch(void* const* d_in, const int* in_sizes, int n_in,
                              void* d_out, int out_size, void* d_ws, size_t ws_size,
                              hipStream_t stream) {
    const float* feat_s1 = (const float*)d_in[0];
    const float* feat_f1 = (const float*)d_in[1];
    const float* feat_c1 = (const float*)d_in[2];
    const int* src_intra = (const int*)d_in[3];
    const int* dst_intra = (const int*)d_in[4];
    const int* src_inter = (const int*)d_in[5];
    const int* dst_inter = (const int*)d_in[6];

    const int NSn = in_sizes[0] / DD;   // 49152
    const int NFn = in_sizes[1] / DD;   // 16384
    const int NCn = in_sizes[2] / DD;   // 4096
    const int E   = in_sizes[3];        // 150000
    const int Bn  = NSn / 48;           // 1024

    // param accessor: leaf-flattened (n_in==27) or per-tuple blobs (n_in==11)
    auto param = [&](int g, int j) -> const float* {
        if (n_in >= 27) return (const float*)d_in[7 + 5 * g + j];
        static const size_t off[5] = {0, 32768, 33280, 33792, 34304}; // W,al,ar,b,Wres
        return (const float*)d_in[7 + g] + off[j];
    };

    // ---- workspace carving (256B aligned) ----
    char* wsp = (char*)d_ws;
    auto carve = [&](size_t bytes) -> char* {
        char* p = wsp; wsp += (bytes + 255) & ~(size_t)255; return p;
    };
    _Float16* xh = (_Float16*)carve((size_t)NSn * DD * 2);
    _Float16* wt[4];
    for (int g = 0; g < 4; ++g) wt[g] = (_Float16*)carve((size_t)NCOL * DD * 2);
    _Float16* wres_sum = (_Float16*)carve((size_t)NCOL * DD * 2);
    float*    hbuf    = (float*)carve((size_t)NSn * NCOL * 4);
    float*    acc     = (float*)carve((size_t)NSn * NCOL * 4);
    float*    el      = (float*)carve((size_t)NSn * HH * 4);
    float*    er      = (float*)carve((size_t)NSn * HH * 4);
    unsigned* menc    = (unsigned*)carve((size_t)NSn * HH * 4);
    float*    mval    = (float*)carve((size_t)NSn * HH * 4);
    float*    sbuf    = (float*)carve((size_t)NSn * HH * 4);
    float*    mean_s1 = (float*)carve((size_t)Bn * DD * 4);
    float*    bsum    = (float*)carve((size_t)NCOL * 4);
    (void)ws_size;

    const int gemmBlocks = (NSn / 16) * (NCOL / 16) / 8;   // 8 waves/block
    const int cvtBlocks  = (64 * NCOL + BT - 1) / BT;

    // ---- converts ----
    k_cvt_f16<<<(NSn * DD + BT - 1) / BT, BT, 0, stream>>>(feat_s1, xh, NSn * DD);
    for (int g = 0; g < 4; ++g)
        k_cvt_transpose<<<cvtBlocks, BT, 0, stream>>>(param(g, 0), wt[g]);
    k_cvt_transpose_sum4<<<cvtBlocks, BT, 0, stream>>>(param(0, 4), param(1, 4),
                                                       param(2, 4), param(3, 4), wres_sum);

    // ---- fused residual GEMM: acc = x @ (sum_g Wres_g) ----
    k_gemm_wmma<<<gemmBlocks, BT, 0, stream>>>(xh, wres_sum, acc, NSn);

    // ---- four GAT relations: p1_intra, p1_inter, p2_intra(rev), p2_inter(rev) ----
    const int* Us[4] = {src_intra, src_inter, dst_intra, dst_inter};
    const int* Vs[4] = {dst_intra, dst_inter, src_intra, src_inter};
    const int nEH  = E * HH;
    const int nNH  = NSn * HH;
    for (int g = 0; g < 4; ++g) {
        k_gemm_wmma<<<gemmBlocks, BT, 0, stream>>>(xh, wt[g], hbuf, NSn);
        k_attn_coef<<<(nNH + BT - 1) / BT, BT, 0, stream>>>(hbuf, param(g, 1), param(g, 2),
                                                            el, er, NSn);
        k_init_ms<<<(nNH + BT - 1) / BT, BT, 0, stream>>>(menc, sbuf, nNH);
        k_edge_max<<<(nEH + BT - 1) / BT, BT, 0, stream>>>(Us[g], Vs[g], el, er, menc, E);
        k_decode_m<<<(nNH + BT - 1) / BT, BT, 0, stream>>>(menc, mval, nNH);
        k_edge_sum<<<(nEH + BT - 1) / BT, BT, 0, stream>>>(Us[g], Vs[g], el, er, mval, sbuf, E);
        k_edge_scatter<<<(nEH + 7) / 8, BT, 0, stream>>>(Us[g], Vs[g], el, er, mval, sbuf,
                                                         hbuf, acc, E);
    }

    // ---- outputs: (h_f1, h_c1, h_s1) concatenated ----
    float* out_f1 = (float*)d_out;
    float* out_c1 = out_f1 + (size_t)NFn * DD;
    float* out_s1 = out_c1 + (size_t)NCn * DD;
    (void)out_size;

    k_bcast_mean<<<(Bn * DD + BT - 1) / BT, BT, 0, stream>>>(feat_f1, out_f1, Bn, NFn / Bn);
    k_bcast_mean<<<(Bn * DD + BT - 1) / BT, BT, 0, stream>>>(feat_c1, out_c1, Bn, NCn / Bn);
    k_seg_mean<<<(Bn * DD + BT - 1) / BT, BT, 0, stream>>>(feat_s1, mean_s1, Bn, NSn / Bn);
    k_bias_sum<<<(NCOL + BT - 1) / BT, BT, 0, stream>>>(param(0, 3), param(1, 3),
                                                        param(2, 3), param(3, 3), bsum);
    k_finalize_s1<<<(NSn * DD + BT - 1) / BT, BT, 0, stream>>>(acc, bsum, mean_s1,
                                                               out_s1, NSn, NSn / Bn);
}
